// SingleHeadAttention_3100966388184
// MI455X (gfx1250) — matrix operand order repaired and checked by reading, not compiled, not run on hardware
//
#include <hip/hip_runtime.h>
#include <hip/hip_bf16.h>

// ---------------------------------------------------------------- constants
constexpr int kD = 1024;    // model dim
constexpr int kB = 4;       // batch
constexpr int kS = 2048;    // new sequence
constexpr int kP = 2048;    // past length
constexpr int kT = kP + kS; // total keys = 4096

typedef __attribute__((ext_vector_type(16))) __bf16          v16bf;
typedef __attribute__((ext_vector_type(8)))  float           v8f;
typedef __attribute__((ext_vector_type(16))) unsigned short  u16x16;
typedef __attribute__((ext_vector_type(4)))  unsigned int    u32x4;
typedef __attribute__((ext_vector_type(8)))  int             i32x8;
typedef __attribute__((ext_vector_type(4)))  int             i32x4;
typedef int v4i32 __attribute__((vector_size(16)));   // matches builtin param

#define GLOBAL_AS __attribute__((address_space(1)))
#define LDS_AS    __attribute__((address_space(3)))

#define HAS_TDM         __has_builtin(__builtin_amdgcn_tensor_load_to_lds)
#define HAS_ASYNC_B128  __has_builtin(__builtin_amdgcn_global_load_async_to_lds_b128)
#define HAS_WAIT_TENSOR __has_builtin(__builtin_amdgcn_s_wait_tensorcnt)
#define HAS_WAIT_ASYNC  __has_builtin(__builtin_amdgcn_s_wait_asynccnt)

// ---------------------------------------------------------------- helpers
__device__ __forceinline__ unsigned short f2bf(float f) {
  unsigned int u = __float_as_uint(f);
  unsigned int r = u + 0x7FFFu + ((u >> 16) & 1u);   // round-to-nearest-even
  return (unsigned short)(r >> 16);
}

__device__ __forceinline__ v8f zero8() {
  v8f z = {0.f, 0.f, 0.f, 0.f, 0.f, 0.f, 0.f, 0.f};
  return z;
}

__device__ __forceinline__ v8f wmma_bf16(v16bf a, v16bf b, v8f c) {
  return __builtin_amdgcn_wmma_f32_16x16x32_bf16(
      false, a, false, b, (short)0, c, false, false);
}

__device__ __forceinline__ unsigned lds_addr_of(const void* p) {
  return (unsigned)(uintptr_t)p;   // generic LDS ptr low 32 bits = LDS offset
}

#if HAS_ASYNC_B128
// async 16B global -> LDS copy, tracked on ASYNCcnt (no VGPR data round-trip)
#define ASYNC_CP_B128(gsrc, ldst)                                   \
  __builtin_amdgcn_global_load_async_to_lds_b128(                   \
      (GLOBAL_AS v4i32*)(uintptr_t)(gsrc),                          \
      (LDS_AS v4i32*)(unsigned)(uintptr_t)(ldst), 0, 0)

__device__ __forceinline__ void wait_async0() {
#if HAS_WAIT_ASYNC
  __builtin_amdgcn_s_wait_asynccnt(0);
#else
  asm volatile("s_wait_asynccnt 0x0" ::: "memory");
#endif
}
#endif

#if HAS_TDM
// TDM: DMA a 2-D bf16 tile [rows=32 x cols=1024], row stride 1024 elems,
// from global memory into LDS at lds_byte_addr. D# layout per CDNA5 ISA ch.8.
// This toolchain exposes the 6-arg builtin: (g0, g1, g2, g3, g_extra, cpol).
__device__ __forceinline__ void tdm_load_tile_32x1024(
    unsigned lds_byte_addr, const unsigned short* gptr) {
  unsigned long long ga = (unsigned long long)(uintptr_t)gptr;
  u32x4 g0 = {
      1u,                                   // count=1 valid descriptor
      lds_byte_addr,                        // lds_addr
      (unsigned)(ga & 0xFFFFFFFFull),       // global_addr[31:0]
      (unsigned)((ga >> 32) & 0x01FFFFFFu) | (2u << 30)  // addr[56:32] | type=2
  };
  // group1: data_size=2B(code1); tensor_dim0=1024; tensor_dim1=big;
  //         tile_dim0=1024, tile_dim1=32, tile_dim2=0; dim0_stride=1024.
  i32x8 g1 = {
      (int)(1u << 16),       // w0: workgroup_mask=0 | data_size=1 (2 bytes)
      (int)(1024u << 16),    // w1: atomic_barrier_addr=0 | tensor_dim0[15:0]
      0,                     // w2: tensor_dim0[31:16]=0 | tensor_dim1[15:0]=0
      (int)(0x4000u | (1024u << 16)),  // w3: tensor_dim1[31:16] | tile_dim0=1024
      32,                    // w4: tile_dim1=32 | tile_dim2=0
      1024,                  // w5: tensor_dim0_stride[31:0]
      0,                     // w6: stride0[47:32]=0 | stride1[15:0]=0
      0                      // w7
  };
  i32x4 z4 = {0, 0, 0, 0};
  i32x8 z8 = {0, 0, 0, 0, 0, 0, 0, 0};
  __builtin_amdgcn_tensor_load_to_lds(g0, g1, z4, z4, z8, 0);
}
#endif

// ---------------------------------------------------------------- prep kernels
__global__ void sha_cvt_bf16(const float* __restrict__ src,
                             unsigned short* __restrict__ dst, long n) {
  for (long i = (long)blockIdx.x * blockDim.x + threadIdx.x; i < n;
       i += (long)gridDim.x * blockDim.x)
    dst[i] = f2bf(src[i]);
}

__global__ void sha_copy_past(const float* __restrict__ pk,
                              const float* __restrict__ pv,
                              float* __restrict__ kF, float* __restrict__ vF,
                              unsigned short* __restrict__ kBW,
                              unsigned short* __restrict__ vBW) {
  const size_t n = (size_t)kB * kP * kD;
  for (size_t i = (size_t)blockIdx.x * blockDim.x + threadIdx.x; i < n;
       i += (size_t)gridDim.x * blockDim.x) {
    size_t b = i / ((size_t)kP * kD);
    size_t r = i % ((size_t)kP * kD);
    size_t o = b * (size_t)kT * kD + r;
    float kvv = pk[i], vvv = pv[i];
    kF[o] = kvv;  vF[o] = vvv;
    kBW[o] = f2bf(kvv);  vBW[o] = f2bf(vvv);
  }
}

// ---------------------------------------------------------------- GEMM: C = A @ W^T
// BM=128 x BN=64 block tile; 8 waves, wave w owns rows [16w,16w+16), all 64 cols.
// 4 WMMA per wave per BK=32 stage (A-fragment reused across 4 B-fragments).
__global__ __launch_bounds__(256) void sha_gemm_xw(
    const unsigned short* __restrict__ A, const unsigned short* __restrict__ W,
    float* __restrict__ outF, long fbs, int frow,
    unsigned short* __restrict__ outB, long bbs, int brow) {
  constexpr int BM = 128, BN = 64, BK = 32;
  __shared__ unsigned short As[BM][BK + 8];
  __shared__ unsigned short Bs[BN][BK + 8];

  const int tid  = threadIdx.x;
  const int bm   = blockIdx.x * BM;
  const int bn   = blockIdx.y * BN;
  const int w    = tid >> 5, lane = tid & 31;
  const int hl   = lane >> 4, r = lane & 15;
  const int wm   = w * 16;

  v8f acc[4];
#pragma unroll
  for (int i = 0; i < 4; ++i) acc[i] = zero8();

  for (int k0 = 0; k0 < kD; k0 += BK) {
    // stage A (128x32: 512 16B-chunks) and W (64x32: 256 chunks)
#if HAS_ASYNC_B128
#pragma unroll
    for (int c = 0; c < 2; ++c) {
      int id = tid + c * 256;
      int row = id >> 2, ch = (id & 3) * 8;
      ASYNC_CP_B128(A + (size_t)(bm + row) * kD + k0 + ch, &As[row][ch]);
    }
    {
      int row = tid >> 2, ch = (tid & 3) * 8;
      ASYNC_CP_B128(W + (size_t)(bn + row) * kD + k0 + ch, &Bs[row][ch]);
    }
#else
#pragma unroll
    for (int c = 0; c < 2; ++c) {
      int id = tid + c * 256;
      int row = id >> 2, ch = (id & 3) * 8;
      *(uint4*)(&As[row][ch]) =
          *(const uint4*)(A + (size_t)(bm + row) * kD + k0 + ch);
    }
    {
      int row = tid >> 2, ch = (tid & 3) * 8;
      *(uint4*)(&Bs[row][ch]) =
          *(const uint4*)(W + (size_t)(bn + row) * kD + k0 + ch);
    }
#endif
    if (k0 + BK < kD) {   // advisory prefetch of next slab -> global_prefetch
      __builtin_prefetch(A + (size_t)(bm + (tid >> 1)) * kD + k0 + BK, 0, 3);
      __builtin_prefetch(W + (size_t)(bn + (tid >> 2)) * kD + k0 + BK, 0, 3);
    }
#if HAS_ASYNC_B128
    wait_async0();
#endif
    __syncthreads();

    u16x16 ua;
#pragma unroll
    for (int e = 0; e < 16; ++e) {
      int ka = ((e < 8) ? e : e + 8) + 8 * hl;   // A frag: interleaved K halves
      ua[e] = As[wm + r][ka];
    }
    v16bf fa = __builtin_bit_cast(v16bf, ua);
#pragma unroll
    for (int nt = 0; nt < 4; ++nt) {
      u16x16 ub;
#pragma unroll
      for (int e = 0; e < 16; ++e)
        ub[e] = Bs[nt * 16 + r][((e < 8) ? e : e + 8) + 8 * hl];    // B frag: interleaved K halves
      acc[nt] = wmma_bf16(fa, __builtin_bit_cast(v16bf, ub), acc[nt]);
    }
    __syncthreads();
  }

#pragma unroll
  for (int i = 0; i < 8; ++i) {
    int m  = i + 8 * hl;
    int gm = bm + wm + m;
    int b  = gm / kS, s = gm % kS;
#pragma unroll
    for (int nt = 0; nt < 4; ++nt) {
      int n0 = bn + nt * 16 + r;
      if (outF)
        outF[(size_t)b * fbs + (size_t)(frow + s) * kD + n0] = acc[nt][i];
      if (outB)
        outB[(size_t)b * bbs + (size_t)(brow + s) * kD + n0] = f2bf(acc[nt][i]);
    }
  }
}

// ---------------------------------------------------------------- flash attention
__global__ __launch_bounds__(256) void sha_flash(
    const unsigned short* __restrict__ Q,   // [kB*kS, kD] bf16
    const unsigned short* __restrict__ K,   // [kB*kT, kD] bf16
    const unsigned short* __restrict__ V,   // [kB*kT, kD] bf16
    unsigned short* __restrict__ O) {       // [kB*kS, kD] bf16 context
  __shared__ unsigned short Qs[16][kD];     // 32 KB
  __shared__ unsigned short Ks[32][kD];     // 64 KB
  __shared__ unsigned short Vs[32][kD];     // 64 KB
  __shared__ float Sred[8][16][32];         // 16 KB partial scores
  __shared__ float Sfull[16][32];
  __shared__ float Mrow[16], Lrow[16], Arow[16];

  const int tid  = threadIdx.x;
  const int w    = tid >> 5, lane = tid & 31;
  const int hl   = lane >> 4, r = lane & 15;
  const int blk  = blockIdx.x;
  const int b    = blk >> 7;
  const int qt   = blk & 127;
  const int qb0  = qt * 16;
  const int dsl  = w * 128;

  // one-time Q tile staging
#if HAS_ASYNC_B128
  for (int i = tid; i < 16 * (kD / 8); i += 256) {
    int row = i >> 7, ch = (i & 127) * 8;
    ASYNC_CP_B128(Q + (size_t)(b * kS + qb0 + row) * kD + ch, &Qs[row][ch]);
  }
  if (tid < 16) { Mrow[tid] = -3.0e38f; Lrow[tid] = 0.f; }
  wait_async0();
  __syncthreads();
#else
  for (int i = tid; i < 16 * (kD / 8); i += 256) {
    int row = i >> 7, ch = (i & 127) * 8;
    *(uint4*)(&Qs[row][ch]) =
        *(const uint4*)(Q + (size_t)(b * kS + qb0 + row) * kD + ch);
  }
  if (tid < 16) { Mrow[tid] = -3.0e38f; Lrow[tid] = 0.f; }
  __syncthreads();
#endif

  v8f oacc[8];
#pragma unroll
  for (int i = 0; i < 8; ++i) oacc[i] = zero8();

  const int kmax   = kP + qb0 + 16;
  const int nsteps = (kmax + 31) / 32;
  const float scale = 0.03125f;     // 1/sqrt(1024)

  for (int step = 0; step < nsteps; ++step) {
    const int kbase = step * 32;

    // ---- stage K,V 32x1024 tiles into LDS: TDM > async-copy > plain loads
#if HAS_TDM
    if (w == 0) {
      tdm_load_tile_32x1024(lds_addr_of(&Ks[0][0]),
                            K + (size_t)(b * kT + kbase) * kD);
      tdm_load_tile_32x1024(lds_addr_of(&Vs[0][0]),
                            V + (size_t)(b * kT + kbase) * kD);
#if HAS_WAIT_TENSOR
      __builtin_amdgcn_s_wait_tensorcnt(0);
#else
      asm volatile("s_wait_tensorcnt 0x0" ::: "memory");
#endif
    }
    __syncthreads();
#elif HAS_ASYNC_B128
    for (int i = tid; i < 32 * (kD / 8); i += 256) {
      int row = i >> 7, ch = (i & 127) * 8;
      size_t g = (size_t)(b * kT + kbase + row) * kD + ch;
      ASYNC_CP_B128(K + g, &Ks[row][ch]);
      ASYNC_CP_B128(V + g, &Vs[row][ch]);
    }
    wait_async0();
    __syncthreads();
#else
    for (int i = tid; i < 32 * (kD / 8); i += 256) {
      int row = i >> 7, ch = (i & 127) * 8;
      size_t g = (size_t)(b * kT + kbase + row) * kD + ch;
      *(uint4*)(&Ks[row][ch]) = *(const uint4*)(K + g);
      *(uint4*)(&Vs[row][ch]) = *(const uint4*)(V + g);
    }
    __syncthreads();
#endif

    // ---- phase 1: partial S(16q x 32k) over this wave's 128 d's (8 WMMA)
    v8f s0 = zero8(), s1 = zero8();
#pragma unroll
    for (int c = 0; c < 4; ++c) {
      int d0 = dsl + c * 32;
      u16x16 ua, ub0, ub1;
#pragma unroll
      for (int e = 0; e < 16; ++e) {
        int ka = ((e < 8) ? e : e + 8) + 8 * hl;
        ua[e] = Qs[r][d0 + ka];
        int kk = ((e < 8) ? e : e + 8) + 8 * hl;
        ub0[e] = Ks[r][d0 + kk];
        ub1[e] = Ks[16 + r][d0 + kk];
      }
      v16bf fa = __builtin_bit_cast(v16bf, ua);
      s0 = wmma_bf16(fa, __builtin_bit_cast(v16bf, ub0), s0);
      s1 = wmma_bf16(fa, __builtin_bit_cast(v16bf, ub1), s1);
    }
#pragma unroll
    for (int i = 0; i < 8; ++i) {
      Sred[w][i + 8 * hl][r]      = s0[i];
      Sred[w][i + 8 * hl][16 + r] = s1[i];
    }
    __syncthreads();

    // ---- phase 2: reduce partials, scale, causal mask
    for (int idx = tid; idx < 512; idx += 256) {
      int row = idx >> 5, col = idx & 31;
      float a = 0.f;
#pragma unroll
      for (int ww = 0; ww < 8; ++ww) a += Sred[ww][row][col];
      int j  = kbase + col;
      int iq = qb0 + row;
      Sfull[row][col] = (j > kP + iq) ? -1.0e30f : a * scale;
    }
    __syncthreads();

    // ---- phase 3: online softmax; build P fragment (A layout)
    float m_old = Mrow[r];
    float rowv[32];
    float m_new = m_old;
#pragma unroll
    for (int c2 = 0; c2 < 32; ++c2) {
      rowv[c2] = Sfull[r][c2];
      m_new = fmaxf(m_new, rowv[c2]);
    }
    float alpha = __expf(m_old - m_new);
    float psum = 0.f;
#pragma unroll
    for (int c2 = 0; c2 < 32; ++c2) {
      rowv[c2] = __expf(rowv[c2] - m_new);
      psum += rowv[c2];
    }
    u16x16 up;
#pragma unroll
    for (int e = 0; e < 16; ++e) {
      int kk = ((e < 8) ? e : e + 8) + 8 * hl;
      up[e] = f2bf(rowv[kk]);
    }
    v16bf fp = __builtin_bit_cast(v16bf, up);
    __syncthreads();
    if (w == 0 && lane < 16) {
      Mrow[r] = m_new;
      Lrow[r] = Lrow[r] * alpha + psum;
      Arow[r] = alpha;
    }
    __syncthreads();

    // ---- phase 4: O <- O*alpha + P @ V over 8 d-tiles (8 WMMA)
    float al[8];
#pragma unroll
    for (int i = 0; i < 8; ++i) al[i] = Arow[i + 8 * hl];
#pragma unroll
    for (int dt = 0; dt < 8; ++dt) {
      int dcol = dsl + dt * 16 + r;
      u16x16 uv;
#pragma unroll
      for (int e = 0; e < 16; ++e)
        uv[e] = Vs[((e < 8) ? e : e + 8) + 8 * hl][dcol];
      v8f c = oacc[dt];
#pragma unroll
      for (int i = 0; i < 8; ++i) c[i] *= al[i];
      oacc[dt] = wmma_bf16(fp, __builtin_bit_cast(v16bf, uv), c);
    }
    __syncthreads();
  }

  float li[8];
#pragma unroll
  for (int i = 0; i < 8; ++i) li[i] = 1.0f / Lrow[i + 8 * hl];
#pragma unroll
  for (int dt = 0; dt < 8; ++dt) {
#pragma unroll
    for (int i = 0; i < 8; ++i) {
      int m = i + 8 * hl;
      int dcol = dsl + dt * 16 + r;
      O[(size_t)(b * kS + qb0 + m) * kD + dcol] = f2bf(oacc[dt][i] * li[i]);
    }
  }
}

// ---------------------------------------------------------------- launcher
extern "C" void kernel_launch(void* const* d_in, const int* in_sizes, int n_in,
                              void* d_out, int out_size, void* d_ws, size_t ws_size,
                              hipStream_t stream) {
  const float* x  = (const float*)d_in[0];
  const float* pk = (const float*)d_in[1];
  const float* pv = (const float*)d_in[2];
  const float* Wq = (const float*)d_in[3];
  const float* Wk = (const float*)d_in[4];
  const float* Wv = (const float*)d_in[5];
  const float* Wo = (const float*)d_in[6];
  float* out = (float*)d_out;

  const size_t N_X  = (size_t)kB * kS * kD;
  const size_t N_W  = (size_t)kD * kD;
  const size_t N_KV = (size_t)kB * kT * kD;

  unsigned short* wsp  = (unsigned short*)d_ws;
  unsigned short* xb   = wsp;
  unsigned short* wqb  = xb + N_X;
  unsigned short* wkb  = wqb + N_W;
  unsigned short* wvb  = wkb + N_W;
  unsigned short* wob  = wvb + N_W;
  unsigned short* qbuf = wob + N_W;
  unsigned short* kbuf = qbuf + N_X;
  unsigned short* vbuf = kbuf + N_KV;
  unsigned short* ctxb = vbuf + N_KV;

  float* kF = out + N_X;
  float* vF = kF + N_KV;

  dim3 blk(256);

  sha_cvt_bf16<<<4096, blk, 0, stream>>>(x,  xb,  (long)N_X);
  sha_cvt_bf16<<<1024, blk, 0, stream>>>(Wq, wqb, (long)N_W);
  sha_cvt_bf16<<<1024, blk, 0, stream>>>(Wk, wkb, (long)N_W);
  sha_cvt_bf16<<<1024, blk, 0, stream>>>(Wv, wvb, (long)N_W);
  sha_cvt_bf16<<<1024, blk, 0, stream>>>(Wo, wob, (long)N_W);

  sha_copy_past<<<4096, blk, 0, stream>>>(pk, pv, kF, vF, kbuf, vbuf);

  dim3 ggrid(kB * kS / 128, kD / 64);   // 64 x 16
  sha_gemm_xw<<<ggrid, blk, 0, stream>>>(xb, wqb, nullptr, 0, 0,
                                         qbuf, (long)kS * kD, 0);
  sha_gemm_xw<<<ggrid, blk, 0, stream>>>(xb, wkb, kF, (long)kT * kD, kP,
                                         kbuf, (long)kT * kD, kP);
  sha_gemm_xw<<<ggrid, blk, 0, stream>>>(xb, wvb, vF, (long)kT * kD, kP,
                                         vbuf, (long)kT * kD, kP);

  sha_flash<<<kB * (kS / 16), blk, 0, stream>>>(qbuf, kbuf, vbuf, ctxb);

  sha_gemm_xw<<<ggrid, blk, 0, stream>>>(ctxb, wob, out, (long)kS * kD, 0,
                                         nullptr, 0, 0);
}
